// VectorQuantizer_4449586119189
// MI455X (gfx1250) — matrix operand order repaired and checked by reading, not compiled, not run on hardware
//
#include <hip/hip_runtime.h>

typedef __bf16 bf16_t;
typedef __attribute__((ext_vector_type(16))) __bf16 v16bf;
typedef __attribute__((ext_vector_type(8)))  __bf16 v8bf;
typedef __attribute__((ext_vector_type(8)))  float  v8f;
typedef __attribute__((ext_vector_type(4)))  float  v4f;
typedef __attribute__((ext_vector_type(4)))  int    v4i;

#define LATENT 512
#define NCODES 1024
#define BM 128     // tokens per workgroup
#define BN 128     // codes per inner block
#define KC 32      // K slice per stage
#define LDT 40     // LDS row stride in bf16 elems (32 + 8 pad -> 80B, 16B aligned)

// CDNA5 async global->LDS copy path (ASYNCcnt-tracked), with sync fallback.
#if defined(__HIP_DEVICE_COMPILE__) && defined(__has_builtin)
#if __has_builtin(__builtin_amdgcn_global_load_async_to_lds_b128)
#define VQ_ASYNC 1
#else
#define VQ_ASYNC 0
#endif
#else
#define VQ_ASYNC 0
#endif

#if VQ_ASYNC
typedef __attribute__((address_space(1))) v4i gas_v4i;
typedef __attribute__((address_space(3))) v4i las_v4i;
#endif

__device__ __forceinline__ void copy16_g2lds(const bf16_t* __restrict__ g,
                                             bf16_t* __restrict__ l) {
#if VQ_ASYNC
  __builtin_amdgcn_global_load_async_to_lds_b128(
      (gas_v4i*)g, (las_v4i*)l, 0, 0);
#else
  *(v8bf*)l = *(const v8bf*)g;
#endif
}

__device__ __forceinline__ void wait_async_copies() {
#if VQ_ASYNC
  asm volatile("s_wait_asynccnt 0" ::: "memory");
#endif
}

// ---------------- codebook prep: bf16 hi/lo split + ||c||^2 ----------------
__global__ void vq_prep_codebook(const float* __restrict__ cb,
                                 bf16_t* __restrict__ chi,
                                 bf16_t* __restrict__ clo,
                                 float* __restrict__ b2) {
  __shared__ float red[256];
  const int code = blockIdx.x;
  const float* row = cb + (size_t)code * LATENT;
  float ss = 0.f;
  for (int k = threadIdx.x; k < LATENT; k += 256) {
    float c = row[k];
    bf16_t h = (bf16_t)c;
    bf16_t l = (bf16_t)(c - (float)h);
    chi[(size_t)code * LATENT + k] = h;
    clo[(size_t)code * LATENT + k] = l;
    ss += c * c;
  }
  red[threadIdx.x] = ss;
  __syncthreads();
  for (int s = 128; s > 0; s >>= 1) {
    if (threadIdx.x < s) red[threadIdx.x] += red[threadIdx.x + s];
    __syncthreads();
  }
  if (threadIdx.x == 0) b2[code] = red[0];
}

// ---------------- per-token ||x||^2 (one wave per row) ----------------
__global__ void vq_row_norms(const float* __restrict__ x,
                             float* __restrict__ a2, int ntok) {
  int row = (int)((blockIdx.x * (unsigned)blockDim.x + threadIdx.x) >> 5);
  int lane = threadIdx.x & 31;
  if (row >= ntok) return;
  const float* p = x + (size_t)row * LATENT;
  float ss = 0.f;
  for (int k = lane; k < LATENT; k += 32) { float v = p[k]; ss += v * v; }
  #pragma unroll
  for (int off = 16; off >= 1; off >>= 1) ss += __shfl_xor(ss, off, 32);
  if (lane == 0) a2[row] = ss;
}

// ---------------- main fused GEMM + distance + argmin + gather ----------------
__global__ __launch_bounds__(256)
void vq_main(const float* __restrict__ x,
             const float* __restrict__ cb,
             const bf16_t* __restrict__ chi,
             const bf16_t* __restrict__ clo,
             const float* __restrict__ b2,
             const float* __restrict__ a2,
             float* __restrict__ outq,
             float* __restrict__ outd) {
  __shared__ __align__(16) bf16_t Ahi[BM * LDT];
  __shared__ __align__(16) bf16_t Alo[BM * LDT];
  __shared__ __align__(16) bf16_t Bhi[BN * LDT];
  __shared__ __align__(16) bf16_t Blo[BN * LDT];
  __shared__ float b2s[NCODES];
  __shared__ float a2s[BM];
  __shared__ float redD[BM * 2];
  __shared__ int   redI[BM * 2];
  __shared__ int   bestRow[BM];

  const int tid   = threadIdx.x;
  const int lane  = tid & 31;
  const int l16   = lane & 15;
  const int half  = lane >> 4;
  const int waveId = tid >> 5;
  const int waveM = waveId & 3;   // 0..3 -> 32-row group
  const int waveN = waveId >> 2;  // 0..1 -> 64-col group
  const int mBase = blockIdx.x * BM;

  for (int i = tid; i < NCODES; i += 256) b2s[i] = b2[i];
  for (int i = tid; i < BM; i += 256)     a2s[i] = a2[mBase + i];

  float bestD[2][8];
  int   bestI[2][8];
  #pragma unroll
  for (int tm = 0; tm < 2; ++tm)
    #pragma unroll
    for (int r = 0; r < 8; ++r) { bestD[tm][r] = 3.402823466e38f; bestI[tm][r] = 0; }

  // per-thread staging coordinates
  const int trA = tid >> 3;           // 0..31 (A row group)
  const int tcA = (tid & 7) * 4;      // 0..28 (A col, float4)
  const int trB = tid >> 2;           // 0..63 (B row group)
  const int tcB = (tid & 3) * 8;      // 0,8,16,24 (B col, 8x bf16)

  __syncthreads();

  for (int nb = 0; nb < NCODES / BN; ++nb) {
    v8f acc[2][4];
    #pragma unroll
    for (int tm = 0; tm < 2; ++tm)
      #pragma unroll
      for (int tn = 0; tn < 4; ++tn) {
        v8f z = {0.f, 0.f, 0.f, 0.f, 0.f, 0.f, 0.f, 0.f};
        acc[tm][tn] = z;
      }

    // software-pipeline prologue: preload A slice ks=0 into registers
    v4f aReg[4];
    #pragma unroll
    for (int rp = 0; rp < 4; ++rp)
      aReg[rp] = *(const v4f*)(x + (size_t)(mBase + rp * 32 + trA) * LATENT + tcA);

    for (int ks = 0; ks < LATENT / KC; ++ks) {
      __syncthreads();   // previous stage's compute done reading LDS

      // ---- issue async B copies for this stage (engine writes LDS) ----
      #pragma unroll
      for (int rp = 0; rp < 2; ++rp) {
        int row = rp * 64 + trB;
        size_t g = (size_t)(nb * BN + row) * LATENT + ks * KC + tcB;
        copy16_g2lds(&chi[g], &Bhi[row * LDT + tcB]);
        copy16_g2lds(&clo[g], &Blo[row * LDT + tcB]);
      }

      // ---- convert preloaded A regs -> bf16 hi/lo in LDS ----
      #pragma unroll
      for (int rp = 0; rp < 4; ++rp) {
        int row = rp * 32 + trA;
        #pragma unroll
        for (int j = 0; j < 4; ++j) {
          float c = aReg[rp][j];
          bf16_t h = (bf16_t)c;
          Ahi[row * LDT + tcA + j] = h;
          Alo[row * LDT + tcA + j] = (bf16_t)(c - (float)h);
        }
      }

      // ---- preload next A slice; loads overlap the WMMA burst below ----
      if (ks + 1 < LATENT / KC) {
        #pragma unroll
        for (int rp = 0; rp < 4; ++rp)
          aReg[rp] = *(const v4f*)(x + (size_t)(mBase + rp * 32 + trA) * LATENT +
                                   (ks + 1) * KC + tcA);
      }

      wait_async_copies();
      __syncthreads();

      // ---- A fragments: lanes 0-15 K{0..7,16..23}, lanes 16-31 K{8..15,24..31} ----
      v16bf afh[2], afl[2];
      #pragma unroll
      for (int tm = 0; tm < 2; ++tm) {
        int m = waveM * 32 + tm * 16 + l16;
        int k0 = half * 8;
        v8bf p0h = *(const v8bf*)&Ahi[m * LDT + k0];
        v8bf p1h = *(const v8bf*)&Ahi[m * LDT + 16 + k0];
        v8bf p0l = *(const v8bf*)&Alo[m * LDT + k0];
        v8bf p1l = *(const v8bf*)&Alo[m * LDT + 16 + k0];
        #pragma unroll
        for (int j = 0; j < 8; ++j) {
          afh[tm][j] = p0h[j]; afh[tm][j + 8] = p1h[j];
          afl[tm][j] = p0l[j]; afl[tm][j + 8] = p1l[j];
        }
      }
      // ---- B fragments + 3-product split WMMA ----
      #pragma unroll
      for (int tn = 0; tn < 4; ++tn) {
        int n = waveN * 64 + tn * 16 + l16;
        int k0 = half * 8;
        v8bf q0h = *(const v8bf*)&Bhi[n * LDT + k0];
        v8bf q1h = *(const v8bf*)&Bhi[n * LDT + k0 + 16];
        v8bf q0l = *(const v8bf*)&Blo[n * LDT + k0];
        v8bf q1l = *(const v8bf*)&Blo[n * LDT + k0 + 16];
        v16bf bh, bl;
        #pragma unroll
        for (int j = 0; j < 8; ++j) {
          bh[j] = q0h[j]; bh[j + 8] = q1h[j];
          bl[j] = q0l[j]; bl[j + 8] = q1l[j];
        }
        #pragma unroll
        for (int tm = 0; tm < 2; ++tm) {
          acc[tm][tn] = __builtin_amdgcn_wmma_f32_16x16x32_bf16(
              false, afh[tm], false, bh, (short)0, acc[tm][tn], false, false);
          acc[tm][tn] = __builtin_amdgcn_wmma_f32_16x16x32_bf16(
              false, afl[tm], false, bh, (short)0, acc[tm][tn], false, false);
          acc[tm][tn] = __builtin_amdgcn_wmma_f32_16x16x32_bf16(
              false, afh[tm], false, bl, (short)0, acc[tm][tn], false, false);
        }
      }
    }

    // ---- epilogue for this code block: distances + running argmin ----
    #pragma unroll
    for (int tm = 0; tm < 2; ++tm) {
      int mloc0 = waveM * 32 + tm * 16 + half * 8;
      #pragma unroll
      for (int tn = 0; tn < 4; ++tn) {
        int n = nb * BN + waveN * 64 + tn * 16 + l16;
        float bb = b2s[n];
        size_t base = (size_t)(mBase + mloc0) * NCODES + n;
        #pragma unroll
        for (int r = 0; r < 8; ++r) {
          float d = a2s[mloc0 + r] + bb - 2.0f * acc[tm][tn][r];
          __builtin_nontemporal_store(d, outd + base + (size_t)r * NCODES);
          if (d < bestD[tm][r]) { bestD[tm][r] = d; bestI[tm][r] = n; }
        }
      }
    }
  }

  // ---- cross-lane argmin reduction (16 lanes share a row) ----
  #pragma unroll
  for (int tm = 0; tm < 2; ++tm) {
    #pragma unroll
    for (int r = 0; r < 8; ++r) {
      float d = bestD[tm][r];
      int idx = bestI[tm][r];
      #pragma unroll
      for (int off = 8; off >= 1; off >>= 1) {
        float od = __shfl_xor(d, off, 32);
        int   oi = __shfl_xor(idx, off, 32);
        if (od < d || (od == d && oi < idx)) { d = od; idx = oi; }
      }
      if (l16 == 0) {
        int mloc = waveM * 32 + tm * 16 + half * 8 + r;
        redD[mloc * 2 + waveN] = d;
        redI[mloc * 2 + waveN] = idx;
      }
    }
  }
  __syncthreads();
  if (tid < BM) {
    float d0 = redD[tid * 2], d1 = redD[tid * 2 + 1];
    int   i0 = redI[tid * 2], i1 = redI[tid * 2 + 1];
    bestRow[tid] = (d1 < d0 || (d1 == d0 && i1 < i0)) ? i1 : i0;
  }
  __syncthreads();

  // ---- gather quantized rows (exact f32 codebook copy) ----
  for (int i = tid; i < BM * (LATENT / 4); i += 256) {
    int row  = i >> 7;          // LATENT/4 == 128
    int col4 = i & 127;
    v4f v = *(const v4f*)(cb + (size_t)bestRow[row] * LATENT + col4 * 4);
    __builtin_nontemporal_store(v, (v4f*)(outq + (size_t)(mBase + row) * LATENT + col4 * 4));
  }
}

extern "C" void kernel_launch(void* const* d_in, const int* in_sizes, int n_in,
                              void* d_out, int out_size, void* d_ws, size_t ws_size,
                              hipStream_t stream) {
  const float* x  = (const float*)d_in[0];
  const float* cb = (const float*)d_in[1];
  const int ntok = in_sizes[0] / LATENT;   // 65536

  char* ws = (char*)d_ws;
  bf16_t* chi = (bf16_t*)ws;
  bf16_t* clo = chi + (size_t)NCODES * LATENT;
  float*  b2  = (float*)(clo + (size_t)NCODES * LATENT);
  float*  a2  = b2 + NCODES;

  float* outq = (float*)d_out;
  float* outd = outq + (size_t)ntok * LATENT;

  vq_prep_codebook<<<NCODES, 256, 0, stream>>>(cb, chi, clo, b2);
  vq_row_norms<<<(ntok + 7) / 8, 256, 0, stream>>>(x, a2, ntok);
  vq_main<<<ntok / BM, 256, 0, stream>>>(x, cb, chi, clo, b2, a2, outq, outd);
}
